// SparseMoELayer_65687229825576
// MI455X (gfx1250) — matrix operand order repaired and checked by reading, not compiled, not run on hardware
//
#include <hip/hip_runtime.h>

// ---------------- problem constants ----------------
#define D_MODEL   1024
#define D_FF      4096
#define N_EXPERTS 8
#define NTOK      8192            // 4 * 2048
#define MBLK      128             // GEMM M tile (rows of pair list per block)
#define NBLK      128             // GEMM N tile (8 waves x 16)
#define PAIR_CAP  17408           // 16384 pairs + per-expert pad to 128, mult of 128
#define LDA       40              // LDS A-tile row stride (elements): 80B, 16B aligned, bank-conflict free

// ctrl[] int indices
#define C_CNT   0     // [0..7]   per-expert token counts
#define C_OFF   8     // [8..16]  padded segment offsets (9)
#define C_TOT   17    // total padded pairs
#define C_FILL  18    // [18..25] scatter fill counters
#define C_T1C   26    // [26..33] top-1 counts
#define C_PSUM  34    // [34..41] prob sums (as float)

// CDNA5 async global->LDS copy path (ASYNCcnt), guarded: falls back to
// synchronous VGPR staging if the builtins are absent on this toolchain.
#if defined(__gfx1250__) && __has_builtin(__builtin_amdgcn_global_load_async_to_lds_b128) && \
    __has_builtin(__builtin_amdgcn_s_wait_asynccnt)
#define HAS_ASYNC_LDS 1
#else
#define HAS_ASYNC_LDS 0
#endif

typedef __attribute__((ext_vector_type(16))) __bf16 v16bf;
typedef __attribute__((ext_vector_type(8)))  __bf16 v8bf;
typedef __attribute__((ext_vector_type(8)))  float  v8f;

__device__ inline v16bf pack16(v8bf lo, v8bf hi) {
  v16bf r;
#pragma unroll
  for (int i = 0; i < 8; ++i) { r[i] = lo[i]; r[i + 8] = hi[i]; }
  return r;
}

#if HAS_ASYNC_LDS
// builtin signature (from hipcc diagnostic): params are pointers to
// int __attribute__((vector_size(16))) in global / LDS address spaces.
typedef int v4i32 __attribute__((vector_size(16)));
__device__ inline void async_cp16(const __bf16* g, __bf16* l) {
  // generic->AS casts via uintptr_t: global is identity; LDS generic low 32 bits are the LDS offset
  __builtin_amdgcn_global_load_async_to_lds_b128(
      (__attribute__((address_space(1))) v4i32*)(uintptr_t)g,
      (__attribute__((address_space(3))) v4i32*)(uintptr_t)l, 0, 0);
}
#endif

// ---------------- init: ctrl, pair list (-> zero token), zero row of xb ----------------
__global__ void moe_init_kernel(int* ctrl, int* pair_token, __bf16* xb_zero_row) {
  int i = blockIdx.x * blockDim.x + threadIdx.x;
  if (i < 64) ctrl[i] = 0;
  if (i < PAIR_CAP) pair_token[i] = NTOK;     // pad slots point at the zero row
  if (i < D_MODEL) xb_zero_row[i] = (__bf16)0.0f;
}

// ---------------- f32 -> bf16 elementwise ----------------
__global__ void moe_cvt_bf16_kernel(const float* __restrict__ in, __bf16* __restrict__ out, int n) {
  int i = blockIdx.x * blockDim.x + threadIdx.x;
  if (i < n) out[i] = (__bf16)in[i];
}

// ---------------- (E,R,C) f32 -> (E,C,R) bf16 transpose ----------------
__global__ void moe_transpose_cvt_kernel(const float* __restrict__ in, __bf16* __restrict__ out,
                                         int R, int C) {
  __shared__ float tile[32][33];
  const float* inp = in + (size_t)blockIdx.z * R * C;
  __bf16* outp = out + (size_t)blockIdx.z * R * C;
  int c0 = blockIdx.x * 32, r0 = blockIdx.y * 32;
  int tx = threadIdx.x, ty = threadIdx.y;
  for (int j = ty; j < 32; j += 8) tile[j][tx] = inp[(size_t)(r0 + j) * C + c0 + tx];
  __syncthreads();
  for (int j = ty; j < 32; j += 8) outp[(size_t)(c0 + j) * R + r0 + tx] = (__bf16)tile[tx][j];
}

// ---------------- router: logits, softmax, top-2, stats ----------------
__global__ __launch_bounds__(256) void moe_router_kernel(
    const float* __restrict__ x, const float* __restrict__ gw,
    int* ctrl, int* __restrict__ topk_idx, float* __restrict__ topk_w) {
  __shared__ float slog[N_EXPERTS];
  __shared__ float sprob[N_EXPERTS];
  int n = blockIdx.x;
  int tid = threadIdx.x, wave = tid >> 5, lane = tid & 31;
  const float* xr = x + (size_t)n * D_MODEL;
  float p = 0.f;
  for (int d = lane; d < D_MODEL; d += 32) p += xr[d] * gw[(size_t)d * N_EXPERTS + wave];
#pragma unroll
  for (int off = 16; off; off >>= 1) p += __shfl_xor(p, off, 32);
  if (lane == 0) slog[wave] = p;
  __syncthreads();
  if (tid == 0) {
    float m = slog[0];
#pragma unroll
    for (int e = 1; e < N_EXPERTS; ++e) m = fmaxf(m, slog[e]);
    float pr[N_EXPERTS], s = 0.f;
#pragma unroll
    for (int e = 0; e < N_EXPERTS; ++e) { pr[e] = __expf(slog[e] - m); s += pr[e]; }
    float inv = 1.f / s;
#pragma unroll
    for (int e = 0; e < N_EXPERTS; ++e) { pr[e] *= inv; sprob[e] = pr[e]; }
    int i0 = 0;
#pragma unroll
    for (int e = 1; e < N_EXPERTS; ++e) if (pr[e] > pr[i0]) i0 = e;
    int i1 = (i0 == 0) ? 1 : 0;
#pragma unroll
    for (int e = 0; e < N_EXPERTS; ++e) if (e != i0 && pr[e] > pr[i1]) i1 = e;
    float rs = 1.f / (pr[i0] + pr[i1] + 1e-9f);
    topk_idx[2 * n] = i0;  topk_idx[2 * n + 1] = i1;
    topk_w[2 * n] = pr[i0] * rs;  topk_w[2 * n + 1] = pr[i1] * rs;
    atomicAdd(&ctrl[C_CNT + i0], 1);
    atomicAdd(&ctrl[C_CNT + i1], 1);
    atomicAdd(&ctrl[C_T1C + i0], 1);
  }
  __syncthreads();
  if (tid < N_EXPERTS) atomicAdd(&((float*)ctrl)[C_PSUM + tid], sprob[tid]);
}

// ---------------- padded segment offsets + per-slot expert id ----------------
__global__ void moe_offsets_kernel(int* ctrl, int* __restrict__ pair_expert) {
  __shared__ int soff[N_EXPERTS + 1];
  if (threadIdx.x == 0) {
    int off = 0;
    ctrl[C_OFF + 0] = 0; soff[0] = 0;
    for (int e = 0; e < N_EXPERTS; ++e) {
      int c = ctrl[C_CNT + e];
      off += (c + MBLK - 1) & ~(MBLK - 1);
      ctrl[C_OFF + e + 1] = off; soff[e + 1] = off;
    }
    ctrl[C_TOT] = off;
  }
  __syncthreads();
  for (int s = threadIdx.x; s < PAIR_CAP; s += blockDim.x) {
    int e = 0;
    while (e < N_EXPERTS - 1 && s >= soff[e + 1]) ++e;
    pair_expert[s] = e;
  }
}

// ---------------- scatter tokens into expert segments ----------------
__global__ void moe_scatter_kernel(const int* __restrict__ topk_idx, int* ctrl,
                                   int* __restrict__ pair_token, int* __restrict__ token_pos) {
  int n = blockIdx.x * blockDim.x + threadIdx.x;
  if (n >= NTOK) return;
#pragma unroll
  for (int k = 0; k < 2; ++k) {
    int e = topk_idx[2 * n + k];
    int pos = ctrl[C_OFF + e] + atomicAdd(&ctrl[C_FILL + e], 1);
    pair_token[pos] = n;
    token_pos[2 * n + k] = pos;
  }
}

// ---------------- FFN1: h = silu(x @ w1 + b1), bf16 WMMA, 128x128 tiles ----------------
__global__ __launch_bounds__(256) void moe_ffn1_kernel(
    const __bf16* __restrict__ xb, const __bf16* __restrict__ w1t, const float* __restrict__ b1,
    __bf16* __restrict__ hbuf, const int* __restrict__ pair_token,
    const int* __restrict__ pair_expert, const int* __restrict__ ctrl) {
  __shared__ int srows[MBLK];
#if HAS_ASYNC_LDS
  __shared__ __align__(16) __bf16 As[2][MBLK * LDA];
#else
  __shared__ __align__(16) __bf16 As[1][MBLK * LDA];
#endif
  int m0 = blockIdx.y * MBLK;
  if (m0 >= ctrl[C_TOT]) return;
  int tid = threadIdx.x, wave = tid >> 5, lane = tid & 31;
  if (tid < MBLK) srows[tid] = pair_token[m0 + tid];
  __syncthreads();
  // each thread stages two 16B chunks per K-step (128 rows x 32 cols bf16)
  int row0 = tid >> 2, row1 = row0 + 64, ch = (tid & 3) * 8;
  const __bf16* g0 = xb + (size_t)srows[row0] * D_MODEL + ch;
  const __bf16* g1 = xb + (size_t)srows[row1] * D_MODEL + ch;
  __bf16* l0 = &As[0][row0 * LDA + ch];
  __bf16* l1 = &As[0][row1 * LDA + ch];
  int col = lane & 15;
  int khalfA = (lane & 16) ? 16 : 0;   // A frag: contiguous 16 K at khalf
  int kbB = (lane & 16) ? 16 : 0;      // B frag: contiguous 16 K at kb
  int n0 = blockIdx.x * NBLK + wave * 16;
  int e = pair_expert[m0];
  const __bf16* bp = w1t + ((size_t)e * D_FF + n0 + col) * D_MODEL;
  v8f c[8];
#pragma unroll
  for (int s = 0; s < 8; ++s)
#pragma unroll
    for (int i = 0; i < 8; ++i) c[s][i] = 0.f;
  const int T = D_MODEL / 32;
#if HAS_ASYNC_LDS
  const int BUF = MBLK * LDA;
  async_cp16(g0, l0);
  async_cp16(g1, l1);
  for (int t = 0; t < T; ++t) {
    int cur = t & 1;
    __builtin_amdgcn_s_wait_asynccnt(0);   // this wave's tile-t chunks are in LDS
    __syncthreads();                       // whole block's tile-t chunks are in LDS
    if (t + 1 < T) {                       // overlap next tile's copy with WMMAs
      int nb = cur ^ 1;
      async_cp16(g0 + (t + 1) * 32, l0 + nb * BUF);
      async_cp16(g1 + (t + 1) * 32, l1 + nb * BUF);
    }
    int kk = t * 32;
    const __bf16* base = &As[cur][0];
#else
  for (int t = 0; t < T; ++t) {
    int kk = t * 32;
    v8bf v0 = *(const v8bf*)(g0 + kk);
    v8bf v1 = *(const v8bf*)(g1 + kk);
    __syncthreads();
    *(v8bf*)l0 = v0;
    *(v8bf*)l1 = v1;
    __syncthreads();
    const __bf16* base = &As[0][0];
#endif
    v8bf bl = *(const v8bf*)(bp + kk + kbB);
    v8bf bh = *(const v8bf*)(bp + kk + kbB + 8);
    v16bf bfrag = pack16(bl, bh);
    v16bf afrag[8];
#pragma unroll
    for (int s = 0; s < 8; ++s) {
      const __bf16* ar = base + (s * 16 + col) * LDA + khalfA;
      afrag[s] = pack16(*(const v8bf*)ar, *(const v8bf*)(ar + 8));
    }
#pragma unroll
    for (int s = 0; s < 8; ++s)
      c[s] = __builtin_amdgcn_wmma_f32_16x16x32_bf16(false, afrag[s], false, bfrag,
                                                     (short)0, c[s], false, false);
  }
  float bias = b1[(size_t)e * D_FF + n0 + col];
  int rbase = (lane & 16) ? 8 : 0;
#pragma unroll
  for (int s = 0; s < 8; ++s)
#pragma unroll
    for (int i = 0; i < 8; ++i) {
      float v = c[s][i] + bias;
      float sv = v / (1.f + __expf(-v));   // silu
      hbuf[(size_t)(m0 + s * 16 + rbase + i) * D_FF + n0 + col] = (__bf16)sv;
    }
}

// ---------------- FFN2: y = h @ w2 + b2, bf16 WMMA, 128x128 tiles ----------------
__global__ __launch_bounds__(256) void moe_ffn2_kernel(
    const __bf16* __restrict__ hbuf, const __bf16* __restrict__ w2t, const float* __restrict__ b2,
    float* __restrict__ ybuf, const int* __restrict__ pair_expert, const int* __restrict__ ctrl) {
#if HAS_ASYNC_LDS
  __shared__ __align__(16) __bf16 As[2][MBLK * LDA];
#else
  __shared__ __align__(16) __bf16 As[1][MBLK * LDA];
#endif
  int m0 = blockIdx.y * MBLK;
  if (m0 >= ctrl[C_TOT]) return;
  int tid = threadIdx.x, wave = tid >> 5, lane = tid & 31;
  int row0 = tid >> 2, row1 = row0 + 64, ch = (tid & 3) * 8;
  const __bf16* g0 = hbuf + (size_t)(m0 + row0) * D_FF + ch;
  const __bf16* g1 = hbuf + (size_t)(m0 + row1) * D_FF + ch;
  __bf16* l0 = &As[0][row0 * LDA + ch];
  __bf16* l1 = &As[0][row1 * LDA + ch];
  int col = lane & 15;
  int khalfA = (lane & 16) ? 16 : 0;
  int kbB = (lane & 16) ? 16 : 0;
  int n0 = blockIdx.x * NBLK + wave * 16;
  int e = pair_expert[m0];
  const __bf16* bp = w2t + ((size_t)e * D_MODEL + n0 + col) * D_FF;
  v8f c[8];
#pragma unroll
  for (int s = 0; s < 8; ++s)
#pragma unroll
    for (int i = 0; i < 8; ++i) c[s][i] = 0.f;
  const int T = D_FF / 32;
#if HAS_ASYNC_LDS
  const int BUF = MBLK * LDA;
  async_cp16(g0, l0);
  async_cp16(g1, l1);
  for (int t = 0; t < T; ++t) {
    int cur = t & 1;
    __builtin_amdgcn_s_wait_asynccnt(0);
    __syncthreads();
    if (t + 1 < T) {
      int nb = cur ^ 1;
      async_cp16(g0 + (t + 1) * 32, l0 + nb * BUF);
      async_cp16(g1 + (t + 1) * 32, l1 + nb * BUF);
    }
    int kk = t * 32;
    const __bf16* base = &As[cur][0];
#else
  for (int t = 0; t < T; ++t) {
    int kk = t * 32;
    v8bf v0 = *(const v8bf*)(g0 + kk);
    v8bf v1 = *(const v8bf*)(g1 + kk);
    __syncthreads();
    *(v8bf*)l0 = v0;
    *(v8bf*)l1 = v1;
    __syncthreads();
    const __bf16* base = &As[0][0];
#endif
    v8bf bl = *(const v8bf*)(bp + kk + kbB);
    v8bf bh = *(const v8bf*)(bp + kk + kbB + 8);
    v16bf bfrag = pack16(bl, bh);
    v16bf afrag[8];
#pragma unroll
    for (int s = 0; s < 8; ++s) {
      const __bf16* ar = base + (s * 16 + col) * LDA + khalfA;
      afrag[s] = pack16(*(const v8bf*)ar, *(const v8bf*)(ar + 8));
    }
#pragma unroll
    for (int s = 0; s < 8; ++s)
      c[s] = __builtin_amdgcn_wmma_f32_16x16x32_bf16(false, afrag[s], false, bfrag,
                                                     (short)0, c[s], false, false);
  }
  float bias = b2[(size_t)e * D_MODEL + n0 + col];
  int rbase = (lane & 16) ? 8 : 0;
#pragma unroll
  for (int s = 0; s < 8; ++s)
#pragma unroll
    for (int i = 0; i < 8; ++i)
      ybuf[(size_t)(m0 + s * 16 + rbase + i) * D_MODEL + n0 + col] = c[s][i] + bias;
}

// ---------------- combine: out[n,:] = w0*y[p0,:] + w1*y[p1,:] ----------------
__global__ void moe_combine_kernel(const float* __restrict__ ybuf, const int* __restrict__ token_pos,
                                   const float* __restrict__ topk_w, float* __restrict__ out) {
  int idx = blockIdx.x * blockDim.x + threadIdx.x;
  if (idx >= NTOK * D_MODEL) return;
  int n = idx >> 10, d = idx & (D_MODEL - 1);
  int p0 = token_pos[2 * n], p1 = token_pos[2 * n + 1];
  out[idx] = topk_w[2 * n] * ybuf[(size_t)p0 * D_MODEL + d] +
             topk_w[2 * n + 1] * ybuf[(size_t)p1 * D_MODEL + d];
}

// ---------------- aux loss scalar ----------------
__global__ void moe_aux_kernel(const int* __restrict__ ctrl, float* __restrict__ out) {
  const float* cf = (const float*)ctrl;
  float acc = 0.f;
#pragma unroll
  for (int e = 0; e < N_EXPERTS; ++e)
    acc += ((float)ctrl[C_T1C + e] / (float)NTOK) * (cf[C_PSUM + e] / (float)NTOK);
  out[(size_t)NTOK * D_MODEL] = 0.01f * (float)N_EXPERTS * acc;
}

// ---------------- host launcher ----------------
extern "C" void kernel_launch(void* const* d_in, const int* in_sizes, int n_in,
                              void* d_out, int out_size, void* d_ws, size_t ws_size,
                              hipStream_t stream) {
  const float* x  = (const float*)d_in[0];
  const float* gw = (const float*)d_in[1];
  const float* w1 = (const float*)d_in[2];
  const float* b1 = (const float*)d_in[3];
  const float* w2 = (const float*)d_in[4];
  const float* b2 = (const float*)d_in[5];
  float* out = (float*)d_out;

  char* ws = (char*)d_ws;
  size_t o = 0;
  int*   ctrl        = (int*)(ws + o);  o += 256;
  int*   topk_idx    = (int*)(ws + o);  o += (size_t)NTOK * 2 * 4;
  float* topk_w      = (float*)(ws + o); o += (size_t)NTOK * 2 * 4;
  int*   token_pos   = (int*)(ws + o);  o += (size_t)NTOK * 2 * 4;
  int*   pair_token  = (int*)(ws + o);  o += (size_t)PAIR_CAP * 4;
  int*   pair_expert = (int*)(ws + o);  o += (size_t)PAIR_CAP * 4;
  o = (o + 255) & ~(size_t)255;
  __bf16* xb   = (__bf16*)(ws + o); o += (size_t)(NTOK + 1) * D_MODEL * 2;  // +1 zero row for pads
  o = (o + 255) & ~(size_t)255;
  __bf16* w1t  = (__bf16*)(ws + o); o += (size_t)N_EXPERTS * D_MODEL * D_FF * 2;
  __bf16* w2t  = (__bf16*)(ws + o); o += (size_t)N_EXPERTS * D_MODEL * D_FF * 2;
  __bf16* hbuf = (__bf16*)(ws + o); o += (size_t)PAIR_CAP * D_FF * 2;
  float*  ybuf = (float*)(ws + o);  o += (size_t)PAIR_CAP * D_MODEL * 4;
  if (o > ws_size) return;  // workspace too small; bail deterministically

  // 1) init control block + pair list + zero row
  moe_init_kernel<<<(PAIR_CAP + 255) / 256, 256, 0, stream>>>(ctrl, pair_token,
                                                              xb + (size_t)NTOK * D_MODEL);
  // 2) x -> bf16
  moe_cvt_bf16_kernel<<<(NTOK * D_MODEL) / 256, 256, 0, stream>>>(x, xb, NTOK * D_MODEL);
  // 3) w1 (E,D,F) -> w1t (E,F,D) bf16 ; w2 (E,F,D) -> w2t (E,D,F) bf16
  moe_transpose_cvt_kernel<<<dim3(D_FF / 32, D_MODEL / 32, N_EXPERTS), dim3(32, 8), 0, stream>>>(
      w1, w1t, D_MODEL, D_FF);
  moe_transpose_cvt_kernel<<<dim3(D_MODEL / 32, D_FF / 32, N_EXPERTS), dim3(32, 8), 0, stream>>>(
      w2, w2t, D_FF, D_MODEL);
  // 4) router
  moe_router_kernel<<<NTOK, 256, 0, stream>>>(x, gw, ctrl, topk_idx, topk_w);
  // 5) padded offsets + per-slot expert
  moe_offsets_kernel<<<1, 256, 0, stream>>>(ctrl, pair_expert);
  // 6) scatter
  moe_scatter_kernel<<<NTOK / 256, 256, 0, stream>>>(topk_idx, ctrl, pair_token, token_pos);
  // 7) FFN1 grouped GEMM + silu
  moe_ffn1_kernel<<<dim3(D_FF / NBLK, PAIR_CAP / MBLK), 256, 0, stream>>>(
      xb, w1t, b1, hbuf, pair_token, pair_expert, ctrl);
  // 8) FFN2 grouped GEMM
  moe_ffn2_kernel<<<dim3(D_MODEL / NBLK, PAIR_CAP / MBLK), 256, 0, stream>>>(
      hbuf, w2t, b2, ybuf, pair_expert, ctrl);
  // 9) combine top-2
  moe_combine_kernel<<<(NTOK * D_MODEL) / 256, 256, 0, stream>>>(ybuf, token_pos, topk_w, out);
  // 10) aux loss
  moe_aux_kernel<<<1, 1, 0, stream>>>(ctrl, out);
}
